// LongShortTermMemory_6786048328504
// MI455X (gfx1250) — matrix operand order repaired and checked by reading, not compiled, not run on hardware
//
#include <hip/hip_runtime.h>

// LSTM: BS=512, T=2048, IN=64, HS=128. Batch rows independent in recurrence
// -> 32 workgroups x 16 rows. W/U held as WMMA B-fragments in registers for
// all 2048 steps; x_t/h in LDS; gates via v_wmma_f32_16x16x32_f16; c in regs.
// LDS-only split barriers keep the depth-2 global x prefetch in flight; all
// six A-fragments are batch-loaded before the 12-WMMA burst each step.

typedef _Float16 half_t;
typedef _Float16 v16h __attribute__((ext_vector_type(16)));
typedef _Float16 v8h  __attribute__((ext_vector_type(8)));
typedef _Float16 v4h  __attribute__((ext_vector_type(4)));
typedef float    v8f  __attribute__((ext_vector_type(8)));

#define T_STEPS  2048
#define IN_DIM   64
#define HS       128
#define G4       512
#define ROWS     16
#define NTHREADS 512
#define NBLOCKS  32

#define WSTR 72
#define USTR 136
#define XSTR 72
#define HSTR 136
#define GSTR 520

#define OFF_W  0
#define OFF_U  (OFF_W + G4*WSTR*2)
#define OFF_X  (OFF_U + G4*USTR*2)
#define OFF_H  (OFF_X + ROWS*XSTR*2)
#define OFF_B  (OFF_H + ROWS*HSTR*2)
#define OFF_G  (OFF_B + G4*4)
#define LDS_BYTES (OFF_G + ROWS*GSTR*4)   // 254976 < 320KB WGP LDS

// ---- fast activations: hardware v_tanh_f32 when available ----
#if __has_builtin(__builtin_amdgcn_tanhf)
#define TANHF(x) __builtin_amdgcn_tanhf(x)
#elif __has_builtin(__builtin_amdgcn_tanh_f32)
#define TANHF(x) __builtin_amdgcn_tanh_f32(x)
#else
#define TANHF(x) (2.0f * __builtin_amdgcn_rcpf(1.0f + __expf(-2.0f * (x))) - 1.0f)
#endif
__device__ __forceinline__ float sigmoidf_fast(float x) {
  return 0.5f * TANHF(0.5f * x) + 0.5f;
}

// scheduling fence: keep the batched A-frag loads above the WMMA burst
__device__ __forceinline__ void sched_fence() {
#if __has_builtin(__builtin_amdgcn_sched_barrier)
  __builtin_amdgcn_sched_barrier(0);
#else
  asm volatile("" ::: "memory");
#endif
}

// LDS-only workgroup barrier: order ds ops, but do NOT fence global loads,
// so the streaming x prefetch stays outstanding across barriers.
__device__ __forceinline__ void wg_barrier_lds() {
#if __has_builtin(__builtin_amdgcn_s_wait_dscnt)
  __builtin_amdgcn_s_wait_dscnt(0);
#else
  asm volatile("s_wait_dscnt 0x0" ::: "memory");
#endif
#if __has_builtin(__builtin_amdgcn_s_barrier_signal) && __has_builtin(__builtin_amdgcn_s_barrier_wait)
  __builtin_amdgcn_s_barrier_signal(-1);
  __builtin_amdgcn_s_barrier_wait((short)-1);
#else
  asm volatile("s_barrier_signal -1\n\ts_barrier_wait -1" ::: "memory");
#endif
}

// A-fragment (16x32 f16, ISA 7.12.2): lanes 0-15 hold K kb+{0..7,16..23},
// lanes 16-31 hold K kb+{8..15,24..31}; row = lane%16.
__device__ __forceinline__ v16h load_a_frag(const half_t* row, int kb, int hi) {
  const v8h* plo = (const v8h*)(row + kb + hi * 8);
  const v8h* phi = (const v8h*)(row + kb + 16 + hi * 8);
  v8h lo = *plo;
  v8h hv = *phi;
  return __builtin_shufflevector(lo, hv, 0, 1, 2, 3, 4, 5, 6, 7,
                                 8, 9, 10, 11, 12, 13, 14, 15);
}

// B-fragment (32x16 f16): lanes 0-15 hold K kb+{0..7,16..23}, lanes 16-31 K kb+{8..15,24..31},
// column N = lane%16. col points at K-contiguous storage for column N.
__device__ __forceinline__ v16h load_b_frag(const half_t* col, int kb, int hi) {
  const v8h* p = (const v8h*)(col + kb + hi * 8);
  v8h lo = p[0];
  v8h hv = p[2];
  return __builtin_shufflevector(lo, hv, 0, 1, 2, 3, 4, 5, 6, 7,
                                 8, 9, 10, 11, 12, 13, 14, 15);
}

#define WMMA_F16(a, b, c) \
  __builtin_amdgcn_wmma_f32_16x16x32_f16(false, (a), false, (b), (short)0, (c), false, false)

__global__ __launch_bounds__(NTHREADS, 1)
void lstm_wmma_kernel(const float* __restrict__ x,     // [512,2048,64]
                      const float* __restrict__ W,     // [64,512]
                      const float* __restrict__ U,     // [128,512]
                      const float* __restrict__ bias,  // [512]
                      const float* __restrict__ lw,    // [1,128]
                      const float* __restrict__ lb,    // [1]
                      float* __restrict__ out) {       // y[512] | h[512,128] | c[512,128]
  extern __shared__ char smem[];
  half_t* Wl = (half_t*)(smem + OFF_W);
  half_t* Ul = (half_t*)(smem + OFF_U);
  half_t* xl = (half_t*)(smem + OFF_X);
  half_t* hl = (half_t*)(smem + OFF_H);
  float*  bl = (float*)(smem + OFF_B);
  float*  gl = (float*)(smem + OFF_G);

  const int tid  = threadIdx.x;
  const int wv   = tid >> 5;
  const int lane = tid & 31;
  const int lm   = lane & 15;
  const int hi   = lane >> 4;

  // ---- setup: transpose W,U into LDS as f16, bias, zero h, load x(0) ----
  for (int i = tid; i < IN_DIM * G4; i += NTHREADS) {
    int k = i / G4, n = i % G4;
    Wl[n * WSTR + k] = (half_t)W[i];
  }
  for (int i = tid; i < HS * G4; i += NTHREADS) {
    int k = i / G4, n = i % G4;
    Ul[n * USTR + k] = (half_t)U[i];
  }
  for (int n = tid; n < G4; n += NTHREADS) bl[n] = bias[n];
  for (int i = tid; i < ROWS * HSTR; i += NTHREADS) hl[i] = (half_t)0.0f;

  const int xr = tid >> 5;
  const int xk = (tid & 31) * 2;
  const size_t xrow_base = ((size_t)(blockIdx.x * ROWS + xr) * T_STEPS) * IN_DIM;
  {
    const float2 v = *(const float2*)(x + xrow_base + xk);
    xl[xr * XSTR + xk]     = (half_t)v.x;
    xl[xr * XSTR + xk + 1] = (half_t)v.y;
  }
  __syncthreads();

  const int row = wv;
  const int j0  = lane * 4;
  float creg[4] = {0.0f, 0.0f, 0.0f, 0.0f};

  const float4 bi = *(const float4*)&bl[j0];
  const float4 bf = *(const float4*)&bl[HS + j0];
  const float4 bg = *(const float4*)&bl[2 * HS + j0];
  const float4 bo = *(const float4*)&bl[3 * HS + j0];

  const int n0 = wv * 32 + lm;
  const int n1 = n0 + 16;
  const half_t* arow_x = xl + lm * XSTR;
  const half_t* arow_h = hl + lm * HSTR;
  const int mb = hi * 8;

  // ---- loop-invariant B-fragments (W, U) into registers ----
  v16h bWf[2][2];
#pragma unroll
  for (int kk = 0; kk < 2; ++kk) {
    bWf[kk][0] = load_b_frag(Wl + n0 * WSTR, kk * 32, hi);
    bWf[kk][1] = load_b_frag(Wl + n1 * WSTR, kk * 32, hi);
  }
  v16h bUf[4][2];
#pragma unroll
  for (int kk = 0; kk < 4; ++kk) {
    bUf[kk][0] = load_b_frag(Ul + n0 * USTR, kk * 32, hi);
    bUf[kk][1] = load_b_frag(Ul + n1 * USTR, kk * 32, hi);
  }

  // depth-2 x pipeline: xn_hold carries x(t+1); a load for x(t+2) is issued
  // at the top of iteration t and consumed one full iteration later.
  float2 xn_hold = make_float2(0.0f, 0.0f);
  if (T_STEPS > 1) {
    xn_hold = *(const float2*)(x + xrow_base + (size_t)1 * IN_DIM + xk);
  }

  for (int t = 0; t < T_STEPS; ++t) {
    float2 xn_new = make_float2(0.0f, 0.0f);
    if (t + 2 < T_STEPS) {
      xn_new = *(const float2*)(x + xrow_base + (size_t)(t + 2) * IN_DIM + xk);
    }

    // ---- batch-load ALL six A-fragments into distinct registers ----
    v16h ax[2];
    v16h ah[4];
#pragma unroll
    for (int kk = 0; kk < 2; ++kk) ax[kk] = load_a_frag(arow_x, kk * 32, hi);
#pragma unroll
    for (int kk = 0; kk < 4; ++kk) ah[kk] = load_a_frag(arow_h, kk * 32, hi);
    sched_fence();   // keep the 12 ds_load_b128 above the WMMA burst

    // ---- gates[16,512] = x_t @ W + h @ U : 12-WMMA burst ----
    v8f acc0 = {};
    v8f acc1 = {};
#pragma unroll
    for (int kk = 0; kk < 2; ++kk) {
      acc0 = WMMA_F16(ax[kk], bWf[kk][0], acc0);
      acc1 = WMMA_F16(ax[kk], bWf[kk][1], acc1);
    }
#pragma unroll
    for (int kk = 0; kk < 4; ++kk) {
      acc0 = WMMA_F16(ah[kk], bUf[kk][0], acc0);
      acc1 = WMMA_F16(ah[kk], bUf[kk][1], acc1);
    }
#pragma unroll
    for (int r = 0; r < 8; ++r) {
      gl[(mb + r) * GSTR + n0] = acc0[r];
      gl[(mb + r) * GSTR + n1] = acc1[r];
    }
    wg_barrier_lds();

    // ---- elementwise LSTM update; thread owns (row, j0..j0+3) ----
    const float4 gi = *(const float4*)&gl[row * GSTR + j0];
    const float4 gf = *(const float4*)&gl[row * GSTR + HS + j0];
    const float4 gg = *(const float4*)&gl[row * GSTR + 2 * HS + j0];
    const float4 go = *(const float4*)&gl[row * GSTR + 3 * HS + j0];
    const float iv[4] = {gi.x + bi.x, gi.y + bi.y, gi.z + bi.z, gi.w + bi.w};
    const float fv[4] = {gf.x + bf.x, gf.y + bf.y, gf.z + bf.z, gf.w + bf.w};
    const float gv[4] = {gg.x + bg.x, gg.y + bg.y, gg.z + bg.z, gg.w + bg.w};
    const float ov[4] = {go.x + bo.x, go.y + bo.y, go.z + bo.z, go.w + bo.w};
    v4h hpack;
#pragma unroll
    for (int jj = 0; jj < 4; ++jj) {
      const float it = sigmoidf_fast(iv[jj]);
      const float ft = sigmoidf_fast(fv[jj]);
      const float gt = TANHF(gv[jj]);
      const float ot = sigmoidf_fast(ov[jj]);
      creg[jj] = ft * creg[jj] + it * gt;
      hpack[jj] = (half_t)(ot * TANHF(creg[jj]));
    }
    *(v4h*)&hl[row * HSTR + j0] = hpack;

    // consume x(t+1), loaded one full iteration ago; pin first use here so
    // the loadcnt wait (allowing xn_new to stay in flight) lands at loop bottom
    asm volatile("" : "+v"(xn_hold.x), "+v"(xn_hold.y));
    if (t + 1 < T_STEPS) {
      xl[xr * XSTR + xk]     = (half_t)xn_hold.x;
      xl[xr * XSTR + xk + 1] = (half_t)xn_hold.y;
    }
    xn_hold = xn_new;
    wg_barrier_lds();
  }

  // ---- epilogue: y = h @ lw^T + lb ; emit y[512] | h[512,128] | c[512,128] ----
  const int b = blockIdx.x * ROWS + row;
  float part = 0.0f;
#pragma unroll
  for (int jj = 0; jj < 4; ++jj) {
    const int j = j0 + jj;
    const float hv = (float)hl[row * HSTR + j];
    out[512 + (size_t)b * HS + j]            = hv;
    out[512 + 512 * HS + (size_t)b * HS + j] = creg[jj];
    part += hv * lw[j];
  }
#pragma unroll
  for (int off = 16; off > 0; off >>= 1) part += __shfl_down(part, off, 32);
  if (lane == 0) out[b] = part + lb[0];
}

extern "C" void kernel_launch(void* const* d_in, const int* in_sizes, int n_in,
                              void* d_out, int out_size, void* d_ws, size_t ws_size,
                              hipStream_t stream) {
  const float* x    = (const float*)d_in[0];
  const float* W    = (const float*)d_in[1];
  const float* U    = (const float*)d_in[2];
  const float* bias = (const float*)d_in[3];
  const float* lw   = (const float*)d_in[4];
  const float* lb   = (const float*)d_in[5];
  float* out = (float*)d_out;
  (void)in_sizes; (void)n_in; (void)d_ws; (void)ws_size; (void)out_size;

  lstm_wmma_kernel<<<dim3(NBLOCKS), dim3(NTHREADS), LDS_BYTES, stream>>>(
      x, W, U, bias, lw, lb, out);
}
